// GPT_37933151158717
// MI455X (gfx1250) — compile-verified, matrix operand order checked by reading, not run on hardware
//
#include <hip/hip_runtime.h>
#include <hip/hip_bf16.h>
#include <math.h>

// ---------------------------------------------------------------------------
// GPT-2 small forward (B=4, T=1024, C=768, H=12, L=12, V=50257) for gfx1250.
// GEMMs: V_WMMA_F32_16X16X32_BF16 with double-buffered LDS fed by
// GLOBAL_LOAD_ASYNC_TO_LDS (ASYNCcnt-pipelined); f32->bf16 conversion happens
// at fragment-build time (batched ds_load_b64, then v_cvt_pk_bf16_f32).
// ---------------------------------------------------------------------------

typedef __attribute__((ext_vector_type(16))) __bf16 v16bf;
typedef __attribute__((ext_vector_type(8)))  float  v8f;

namespace cfg {
constexpr int V  = 50257;
constexpr int C  = 768;
constexpr int H  = 12;
constexpr int L  = 12;
constexpr int B  = 4;
constexpr int T  = 1024;
constexpr int HD = C / H;          // 64
constexpr int BT = B * T;          // 4096 rows
constexpr int BTC = BT * C;
}

// ---------------- async global->LDS copy (CDNA5) with fallback -------------

#if defined(__AMDGCN__) && \
    __has_builtin(__builtin_amdgcn_global_load_async_to_lds_b128) && \
    __has_builtin(__builtin_amdgcn_global_load_async_to_lds_b32)
#define HAVE_ASYNC 1
#else
#define HAVE_ASYNC 0
#endif

#if HAVE_ASYNC
typedef int v4i_t __attribute__((vector_size(4 * sizeof(int))));
typedef __attribute__((address_space(1))) v4i_t GlobalV4i;
typedef __attribute__((address_space(3))) v4i_t LdsV4i;
typedef __attribute__((address_space(1))) int   GlobalI32;
typedef __attribute__((address_space(3))) int   LdsI32;
__device__ __forceinline__ void cp_glds16(const float* g, float* l) {
  __builtin_amdgcn_global_load_async_to_lds_b128(
      (GlobalV4i*)g, (LdsV4i*)l, 0, 0);
}
__device__ __forceinline__ void cp_glds4(const float* g, float* l) {
  __builtin_amdgcn_global_load_async_to_lds_b32(
      (GlobalI32*)g, (LdsI32*)l, 0, 0);
}
#if __has_builtin(__builtin_amdgcn_s_wait_asynccnt)
#define WAIT_ASYNC(N) __builtin_amdgcn_s_wait_asynccnt(N)
#else
#define WAIT_ASYNC(N) asm volatile("s_wait_asynccnt %0" ::"i"(N))
#endif
#else
__device__ __forceinline__ void cp_glds16(const float* g, float* l) {
  *(float4*)l = *(const float4*)g;
}
__device__ __forceinline__ void cp_glds4(const float* g, float* l) { *l = *g; }
#define WAIT_ASYNC(N)
#endif

// ------------------------------ GEMM ---------------------------------------
// C[M,N] = act( A[M,K] * op(B) + bias ),  op(B)=B[N,K] row-major if TRANSB
// (X @ W^T with W=[out,in]), else B[K,N] row-major.
// Tile: BM=64 x BN=128 x BK=32, 8 waves (2x4), each wave 32x32 = 2x2 WMMA.
// LDS holds raw f32 tiles (async DMA can't convert); fragments are built with
// ds_load_b64 + v_cvt_pk_bf16_f32 per the CDNA5 16-bit A/B VGPR layout:
//   element e (0..15) of lane -> k = (e&7) + 16*(e>=8) + 8*(lane>=16)

#define GBM 64
#define GBN 128
#define GBK 32
#define LDS_A 34   // f32 stride: 136B rows -> 8B aligned float2 reads
#define LDS_B 34

__device__ __forceinline__ v16bf build_frag_f32(const float* rowbase, int hi) {
  // Phase 1: issue all 8 ds_load_b64 into independent temps so they pipeline
  // behind a single s_wait_dscnt instead of one wait per pair.
  float2 t[8];
  #pragma unroll
  for (int j = 0; j < 8; ++j) {
    int kp = 2 * j + ((j >= 4) ? 8 : 0) + hi * 8;       // even
    t[j] = *(const float2*)(rowbase + kp);
  }
  // Phase 2: pack-convert (v_cvt_pk_bf16_f32).
  union { v16bf v; unsigned u[8]; } f;
  #pragma unroll
  for (int j = 0; j < 8; ++j) {
    union { __bf16 h[2]; unsigned u; } cv;
    cv.h[0] = (__bf16)t[j].x;
    cv.h[1] = (__bf16)t[j].y;
    f.u[j] = cv.u;
  }
  return f.v;
}

template <bool TRANSB, int ACT>  // ACT: 0=none, 1=exact GELU
__global__ __launch_bounds__(256) void gemm_bf16_wmma(
    const float* __restrict__ A, const float* __restrict__ Bm,
    const float* __restrict__ bias, float* __restrict__ Cm,
    int M, int N, int K, int lda, int ldb, int ldc)
{
  __shared__ float Asf[2][GBM * LDS_A];   // [m][k]
  __shared__ float Bsf[2][GBN * LDS_B];   // [n][k] (k-innermost)

  const int tid    = threadIdx.x;
  const int lane   = tid & 31;
  const int wave   = tid >> 5;
  const int wy     = wave >> 2;   // 0..1
  const int wx     = wave & 3;    // 0..3
  const int laneLo = lane & 15;
  const int hi     = lane >> 4;   // 0/1
  const int tileM  = blockIdx.y * GBM;
  const int tileN  = blockIdx.x * GBN;

  // Per-wave ASYNCcnt increments per staged tile (uniform: OOB addrs are
  // clamped, not predicated, so the pipelined wait count is exact).
  constexpr int NOPS = TRANSB ? 6 : 18;   // A: 2x b128; B: 4x b128 or 16x b32

  // staging coordinates (k-invariant)
  const int ar = tid >> 2, ac = (tid & 3) * 8;          // A: 8 f32 / thread
  const int am = min(tileM + ar, M - 1);
  const float* aG = A + (size_t)am * lda + ac;
  const int bn = tid >> 1, bc = (tid & 1) * 16;         // B: 16 f32 / thread
  const int gn = min(tileN + bn, N - 1);                // clamp, see above
  const float* bG = TRANSB ? (Bm + (size_t)gn * ldb + bc)
                           : (Bm + (size_t)bc * ldb + gn);

  v8f acc[2][2];
  v8f zero = {};
  #pragma unroll
  for (int i = 0; i < 2; ++i)
    #pragma unroll
    for (int j = 0; j < 2; ++j) acc[i][j] = zero;

  const int nk = K / GBK;   // all call sites have K % 32 == 0

  // ---- stage tile 0 into buffer 0 ----
  {
    float* aL = &Asf[0][ar * LDS_A + ac];
    cp_glds16(aG, aL); cp_glds16(aG + 4, aL + 4);
    float* bL = &Bsf[0][bn * LDS_B + bc];
    if (TRANSB) {
      cp_glds16(bG, bL);         cp_glds16(bG + 4,  bL + 4);
      cp_glds16(bG + 8, bL + 8); cp_glds16(bG + 12, bL + 12);
    } else {
      #pragma unroll
      for (int e = 0; e < 16; ++e) cp_glds4(bG + (size_t)e * ldb, bL + e);
    }
  }

  for (int kt = 0; kt < nk; ++kt) {
    const int cur = kt & 1;
    if (kt + 1 < nk) {
      // ---- prefetch tile kt+1 into the other buffer ----
      const int k0 = (kt + 1) * GBK;
      float* aL = &Asf[cur ^ 1][ar * LDS_A + ac];
      const float* ag = aG + k0;
      cp_glds16(ag, aL); cp_glds16(ag + 4, aL + 4);
      float* bL = &Bsf[cur ^ 1][bn * LDS_B + bc];
      if (TRANSB) {
        const float* bg = bG + k0;
        cp_glds16(bg, bL);         cp_glds16(bg + 4,  bL + 4);
        cp_glds16(bg + 8, bL + 8); cp_glds16(bg + 12, bL + 12);
      } else {
        const float* bg = bG + (size_t)k0 * ldb;
        #pragma unroll
        for (int e = 0; e < 16; ++e) cp_glds4(bg + (size_t)e * ldb, bL + e);
      }
      WAIT_ASYNC(NOPS);   // current tile done; next tile may be in flight
    } else {
      WAIT_ASYNC(0);
    }
    __syncthreads();

    // ---- fragments (built once, reused across the 2x2 WMMA tiles) ----
    v16bf af[2], bfr[2];
    #pragma unroll
    for (int mi = 0; mi < 2; ++mi)
      af[mi] = build_frag_f32(&Asf[cur][(wy * 32 + mi * 16 + laneLo) * LDS_A], hi);
    #pragma unroll
    for (int ni = 0; ni < 2; ++ni)
      bfr[ni] = build_frag_f32(&Bsf[cur][(wx * 32 + ni * 16 + laneLo) * LDS_B], hi);

    #pragma unroll
    for (int mi = 0; mi < 2; ++mi)
      #pragma unroll
      for (int ni = 0; ni < 2; ++ni)
        acc[mi][ni] = __builtin_amdgcn_wmma_f32_16x16x32_bf16(
            false, af[mi], false, bfr[ni], (short)0, acc[mi][ni], false, false);

    __syncthreads();
  }

  // ---- epilogue: D layout -> lane holds N=lane&15, VGPR r holds M=r+8*hi ----
  #pragma unroll
  for (int mi = 0; mi < 2; ++mi) {
    #pragma unroll
    for (int ni = 0; ni < 2; ++ni) {
      int n = tileN + wx * 32 + ni * 16 + laneLo;
      if (n >= N) continue;
      float bv = bias ? bias[n] : 0.0f;
      #pragma unroll
      for (int r = 0; r < 8; ++r) {
        int m = tileM + wy * 32 + mi * 16 + r + hi * 8;
        if (m >= M) continue;
        float v = acc[mi][ni][r] + bv;
        if (ACT == 1) v = 0.5f * v * (1.0f + erff(v * 0.70710678118654752f));
        Cm[(size_t)m * ldc + n] = v;
      }
    }
  }
}

// --------------------------- elementwise kernels ----------------------------

__global__ __launch_bounds__(256) void embed_kernel(
    const float* __restrict__ wte, const float* __restrict__ wpe,
    const int* __restrict__ idx, float* __restrict__ x)
{
  int i   = blockIdx.x;        // token 0..B*T-1
  int t   = i & (cfg::T - 1);
  int tok = idx[i];
  const float* we = wte + (size_t)tok * cfg::C;
  const float* pe = wpe + (size_t)t * cfg::C;
  float* xr = x + (size_t)i * cfg::C;
  for (int c = threadIdx.x; c < cfg::C; c += 256) xr[c] = we[c] + pe[c];
}

__global__ __launch_bounds__(256) void layernorm_kernel(
    const float* __restrict__ x, const float* __restrict__ w,
    const float* __restrict__ b, float* __restrict__ y)
{
  __shared__ float red[256];
  int row = blockIdx.x;
  const float* xr = x + (size_t)row * cfg::C;
  float s = 0.0f;
  for (int c = threadIdx.x; c < cfg::C; c += 256) s += xr[c];
  red[threadIdx.x] = s; __syncthreads();
  for (int off = 128; off > 0; off >>= 1) {
    if (threadIdx.x < off) red[threadIdx.x] += red[threadIdx.x + off];
    __syncthreads();
  }
  float mu = red[0] * (1.0f / cfg::C);
  __syncthreads();
  float v = 0.0f;
  for (int c = threadIdx.x; c < cfg::C; c += 256) {
    float d = xr[c] - mu; v += d * d;
  }
  red[threadIdx.x] = v; __syncthreads();
  for (int off = 128; off > 0; off >>= 1) {
    if (threadIdx.x < off) red[threadIdx.x] += red[threadIdx.x + off];
    __syncthreads();
  }
  float rstd = rsqrtf(red[0] * (1.0f / cfg::C) + 1e-5f);
  float* yr = y + (size_t)row * cfg::C;
  for (int c = threadIdx.x; c < cfg::C; c += 256)
    yr[c] = (xr[c] - mu) * rstd * w[c] + b[c];
}

__global__ __launch_bounds__(256) void add_kernel(
    float* __restrict__ x, const float* __restrict__ r, int n)
{
  int i = blockIdx.x * 256 + threadIdx.x;
  int stride = gridDim.x * 256;
  for (; i < n; i += stride) x[i] += r[i];
}

// causal softmax over one score row; 1/sqrt(HD) folded in here
__global__ __launch_bounds__(256) void softmax_causal_kernel(
    float* __restrict__ att, float scale)
{
  __shared__ float red[256];
  int m = blockIdx.x;
  float* row = att + (size_t)m * cfg::T;
  int nvalid = m + 1;
  float mx = -1e30f;
  for (int n = threadIdx.x; n < nvalid; n += 256)
    mx = fmaxf(mx, row[n] * scale);
  red[threadIdx.x] = mx; __syncthreads();
  for (int off = 128; off > 0; off >>= 1) {
    if (threadIdx.x < off)
      red[threadIdx.x] = fmaxf(red[threadIdx.x], red[threadIdx.x + off]);
    __syncthreads();
  }
  mx = red[0]; __syncthreads();
  float s = 0.0f;
  for (int n = threadIdx.x; n < nvalid; n += 256) {
    float e = __expf(row[n] * scale - mx);
    row[n] = e; s += e;
  }
  red[threadIdx.x] = s; __syncthreads();
  for (int off = 128; off > 0; off >>= 1) {
    if (threadIdx.x < off) red[threadIdx.x] += red[threadIdx.x + off];
    __syncthreads();
  }
  float inv = 1.0f / red[0];
  for (int n = threadIdx.x; n < cfg::T; n += 256)
    row[n] = (n < nvalid) ? row[n] * inv : 0.0f;
}

// per-row NLL over the vocab dimension of the logits (already in d_out)
__global__ __launch_bounds__(256) void row_nll_kernel(
    const float* __restrict__ logits, const int* __restrict__ targets,
    float* __restrict__ nll)
{
  __shared__ float red[256];
  int row = blockIdx.x;
  const float* lr = logits + (size_t)row * cfg::V;
  float mx = -1e30f;
  for (int v = threadIdx.x; v < cfg::V; v += 256) mx = fmaxf(mx, lr[v]);
  red[threadIdx.x] = mx; __syncthreads();
  for (int off = 128; off > 0; off >>= 1) {
    if (threadIdx.x < off)
      red[threadIdx.x] = fmaxf(red[threadIdx.x], red[threadIdx.x + off]);
    __syncthreads();
  }
  mx = red[0]; __syncthreads();
  float s = 0.0f;
  for (int v = threadIdx.x; v < cfg::V; v += 256) s += __expf(lr[v] - mx);
  red[threadIdx.x] = s; __syncthreads();
  for (int off = 128; off > 0; off >>= 1) {
    if (threadIdx.x < off) red[threadIdx.x] += red[threadIdx.x + off];
    __syncthreads();
  }
  if (threadIdx.x == 0)
    nll[row] = logf(red[0]) + mx - lr[targets[row]];
}

// fixed-order deterministic mean of B*T nll values -> loss scalar
__global__ __launch_bounds__(256) void loss_reduce_kernel(
    const float* __restrict__ nll, float* __restrict__ loss_out)
{
  __shared__ float red[256];
  float s = 0.0f;
  for (int k = 0; k < cfg::BT / 256; ++k) s += nll[threadIdx.x + k * 256];
  red[threadIdx.x] = s; __syncthreads();
  for (int off = 128; off > 0; off >>= 1) {
    if (threadIdx.x < off) red[threadIdx.x] += red[threadIdx.x + off];
    __syncthreads();
  }
  if (threadIdx.x == 0) loss_out[0] = red[0] * (1.0f / cfg::BT);
}

// ------------------------------- launch -------------------------------------

static inline dim3 gemm_grid(int M, int N) {
  return dim3((N + GBN - 1) / GBN, (M + GBM - 1) / GBM);
}

extern "C" void kernel_launch(void* const* d_in, const int* in_sizes, int n_in,
                              void* d_out, int out_size, void* d_ws, size_t ws_size,
                              hipStream_t stream)
{
  using namespace cfg;
  (void)in_sizes; (void)n_in; (void)out_size; (void)ws_size;

  // ---- inputs (setup_inputs dict order, nested dicts flattened) ----
  const float* wte = (const float*)d_in[0];            // [V,C]
  const float* wpe = (const float*)d_in[1];            // [BS,C]
  const float* lnf_w = (const float*)d_in[2 + 12 * L + 0];
  const float* lnf_b = (const float*)d_in[2 + 12 * L + 1];
  const int*   idx     = (const int*)d_in[2 + 12 * L + 2];
  const int*   targets = (const int*)d_in[2 + 12 * L + 3];

  float* out = (float*)d_out;                           // [B*T*V] logits + [1] loss

  // ---- workspace carve-up (floats) ----
  float* ws   = (float*)d_ws;
  float* x    = ws;                 // [B*T, C]   residual stream
  float* ln   = x    + BTC;         // [B*T, C]   layernorm output
  float* qkv  = ln   + BTC;         // [B*T, 3C]
  float* y    = qkv  + 3 * BTC;     // [B*T, C]   attention output (heads merged)
  float* t2   = y    + BTC;         // [B*T, C]   proj / mlp output
  float* hbuf = t2   + BTC;         // [B*T, 4C]  mlp hidden
  float* att  = hbuf + 4 * BTC;     // [T, T]     score/prob buffer (reused per b,h)
  float* nll  = att  + (size_t)T * T; // [B*T]

  const float scale = 1.0f / sqrtf((float)HD); // 0.125

  // ---- embedding ----
  embed_kernel<<<BT, 256, 0, stream>>>(wte, wpe, idx, x);

  // ---- transformer blocks ----
  for (int l = 0; l < L; ++l) {
    const float* ln1_w  = (const float*)d_in[2 + 12 * l + 0];
    const float* ln1_b  = (const float*)d_in[2 + 12 * l + 1];
    const float* attn_w = (const float*)d_in[2 + 12 * l + 2]; // [3C, C]
    const float* attn_b = (const float*)d_in[2 + 12 * l + 3];
    const float* proj_w = (const float*)d_in[2 + 12 * l + 4]; // [C, C]
    const float* proj_b = (const float*)d_in[2 + 12 * l + 5];
    const float* ln2_w  = (const float*)d_in[2 + 12 * l + 6];
    const float* ln2_b  = (const float*)d_in[2 + 12 * l + 7];
    const float* fc_w   = (const float*)d_in[2 + 12 * l + 8]; // [4C, C]
    const float* fc_b   = (const float*)d_in[2 + 12 * l + 9];
    const float* fc2_w  = (const float*)d_in[2 + 12 * l + 10]; // [C, 4C]
    const float* fc2_b  = (const float*)d_in[2 + 12 * l + 11];

    // ln1 + qkv = ln(x) @ attn_w^T + attn_b
    layernorm_kernel<<<BT, 256, 0, stream>>>(x, ln1_w, ln1_b, ln);
    gemm_bf16_wmma<true, 0><<<gemm_grid(BT, 3 * C), 256, 0, stream>>>(
        ln, attn_w, attn_b, qkv, BT, 3 * C, C, C, C, 3 * C);

    // attention per (b,h); score buffer reused (stream-serialized)
    for (int b = 0; b < B; ++b) {
      for (int h = 0; h < H; ++h) {
        const float* qp = qkv + (size_t)b * T * 3 * C + h * HD;
        const float* kp = qp + C;
        const float* vp = qp + 2 * C;
        // scores = q @ k^T  [T,T], K=HD
        gemm_bf16_wmma<true, 0><<<gemm_grid(T, T), 256, 0, stream>>>(
            qp, kp, nullptr, att, T, T, HD, 3 * C, 3 * C, T);
        softmax_causal_kernel<<<T, 256, 0, stream>>>(att, scale);
        // y_slice = probs @ v  [T,HD], K=T
        gemm_bf16_wmma<false, 0><<<gemm_grid(T, HD), 256, 0, stream>>>(
            att, vp, nullptr, y + (size_t)b * T * C + h * HD,
            T, HD, T, T, 3 * C, C);
      }
    }

    // proj + residual
    gemm_bf16_wmma<true, 0><<<gemm_grid(BT, C), 256, 0, stream>>>(
        y, proj_w, proj_b, t2, BT, C, C, C, C, C);
    add_kernel<<<1024, 256, 0, stream>>>(x, t2, BTC);

    // ln2 + MLP (GELU fused in fc epilogue) + residual
    layernorm_kernel<<<BT, 256, 0, stream>>>(x, ln2_w, ln2_b, ln);
    gemm_bf16_wmma<true, 1><<<gemm_grid(BT, 4 * C), 256, 0, stream>>>(
        ln, fc_w, fc_b, hbuf, BT, 4 * C, C, C, C, 4 * C);
    gemm_bf16_wmma<true, 0><<<gemm_grid(BT, C), 256, 0, stream>>>(
        hbuf, fc2_w, fc2_b, t2, BT, C, 4 * C, 4 * C, 4 * C, C);
    add_kernel<<<1024, 256, 0, stream>>>(x, t2, BTC);
  }

  // ---- final LN + tied-weight logits ----
  layernorm_kernel<<<BT, 256, 0, stream>>>(x, lnf_w, lnf_b, ln);
  gemm_bf16_wmma<true, 0><<<gemm_grid(BT, V), 256, 0, stream>>>(
      ln, wte, nullptr, out, BT, V, C, C, C, V);

  // ---- loss = mean NLL (deterministic fixed-order reduction) ----
  row_nll_kernel<<<BT, 256, 0, stream>>>(out, targets, nll);
  loss_reduce_kernel<<<1, 256, 0, stream>>>(nll, out + (size_t)BT * V);
}
